// Struct2Seq_83820581748818
// MI455X (gfx1250) — matrix operand order repaired and checked by reading, not compiled, not run on hardware
//
#include <hip/hip_runtime.h>
#include <hip/hip_bf16.h>
#include <math.h>

typedef _Float16 h16;
typedef __attribute__((ext_vector_type(16))) _Float16 v16h;
typedef __attribute__((ext_vector_type(8)))  _Float16 v8h;
typedef __attribute__((ext_vector_type(8)))  float    v8f;

#define Bdim 8
#define Ndim 1024
#define Hdim 128
#define KNN  30
#define NODES (Bdim*Ndim)          // 8192
#define EDGES (NODES*KNN)          // 245760
#define EIN   39

// ---------------- small vector helpers ----------------
__device__ __forceinline__ void v_unit(float* v, float eps = 1e-8f) {
    float n = sqrtf(v[0]*v[0] + v[1]*v[1] + v[2]*v[2]) + eps;
    v[0] /= n; v[1] /= n; v[2] /= n;
}
__device__ __forceinline__ void v_cross(const float* a, const float* b, float* o) {
    o[0] = a[1]*b[2] - a[2]*b[1];
    o[1] = a[2]*b[0] - a[0]*b[2];
    o[2] = a[0]*b[1] - a[1]*b[0];
}
__device__ __forceinline__ float v_dot(const float* a, const float* b) {
    return a[0]*b[0] + a[1]*b[1] + a[2]*b[2];
}

// ---------------- dihedral node features (B,N,6) ----------------
__global__ void k_dihedrals(const float* __restrict__ X, float* __restrict__ Vfeat)
{
    int t = blockIdx.x * blockDim.x + threadIdx.x;
    if (t >= NODES * 3) return;
    int c = t % 3, node = t / 3;
    int b = node / Ndim, i = node % Ndim;
    int tt = 3 * i + c - 1;                       // index into D_flat
    float D = 0.f;
    if (tt >= 0 && tt <= 3 * Ndim - 4) {
        const float* Xb = X + (size_t)b * Ndim * 4 * 3;
        float p[4][3];
        #pragma unroll
        for (int q = 0; q < 4; q++) {
            int m = tt + q; int res = m / 3, at = m % 3;
            const float* s = Xb + ((size_t)res * 4 + at) * 3;
            p[q][0] = s[0]; p[q][1] = s[1]; p[q][2] = s[2];
        }
        float u2[3], u1[3], u0[3];
        #pragma unroll
        for (int d = 0; d < 3; d++) { u2[d] = p[1][d]-p[0][d]; u1[d] = p[2][d]-p[1][d]; u0[d] = p[3][d]-p[2][d]; }
        v_unit(u2); v_unit(u1); v_unit(u0);
        float n2[3], n1[3];
        v_cross(u2, u1, n2); v_unit(n2);
        v_cross(u1, u0, n1); v_unit(n1);
        float cd = v_dot(n2, n1);
        cd = fminf(fmaxf(cd, -1.f + 1e-7f), 1.f - 1e-7f);
        float s = v_dot(u2, n1);
        float sg = (s > 0.f) ? 1.f : ((s < 0.f) ? -1.f : 0.f);
        D = sg * acosf(cd);
    }
    Vfeat[(size_t)node * 6 + c]     = cosf(D);
    Vfeat[(size_t)node * 6 + 3 + c] = sinf(D);
}

// ---------------- kNN top-30 on CA distances ----------------
__global__ __launch_bounds__(128) void k_topk(const float* __restrict__ X,
                                              const float* __restrict__ mask,
                                              int* __restrict__ eidx,
                                              float* __restrict__ Dnb)
{
    __shared__ float xs[Ndim], ys[Ndim], zs[Ndim], ms[Ndim];
    int b = blockIdx.y, tid = threadIdx.x;
    for (int j = tid; j < Ndim; j += 128) {
        const float* p = X + (((size_t)b * Ndim + j) * 4 + 1) * 3;
        xs[j] = p[0]; ys[j] = p[1]; zs[j] = p[2];
        ms[j] = mask[b * Ndim + j];
    }
    __syncthreads();
    int i = blockIdx.x * 128 + tid;
    float xi = xs[i], yi = ys[i], zi = zs[i], mi = ms[i];
    float maxD = 0.f;
    for (int j = 0; j < Ndim; j++) {
        float dx = xi - xs[j], dy = yi - ys[j], dz = zi - zs[j];
        float d = ms[j] * mi * sqrtf(dx*dx + dy*dy + dz*dz + 1e-6f);
        maxD = fmaxf(maxD, d);
    }
    float key[KNN]; int idx[KNN];
    #pragma unroll
    for (int s = 0; s < KNN; s++) { key[s] = 3.4e38f; idx[s] = 0; }
    for (int j = 0; j < Ndim; j++) {
        float dx = xi - xs[j], dy = yi - ys[j], dz = zi - zs[j];
        float m2 = ms[j] * mi;
        float d = m2 * sqrtf(dx*dx + dy*dy + dz*dz + 1e-6f) + (1.f - m2) * maxD;
        if (d < key[KNN-1]) {
            key[KNN-1] = d; idx[KNN-1] = j;
            #pragma unroll
            for (int s = KNN-1; s > 0; s--) {
                if (key[s] < key[s-1]) {
                    float tk = key[s]; key[s] = key[s-1]; key[s-1] = tk;
                    int ti = idx[s]; idx[s] = idx[s-1]; idx[s-1] = ti;
                }
            }
        }
    }
    size_t base = ((size_t)b * Ndim + i) * KNN;
    #pragma unroll
    for (int s = 0; s < KNN; s++) { eidx[base + s] = idx[s]; Dnb[base + s] = key[s]; }
}

// ---------------- per-node orientation frame O (B,N,9) ----------------
__global__ void k_orient(const float* __restrict__ X, float* __restrict__ O)
{
    int node = blockIdx.x * blockDim.x + threadIdx.x;
    if (node >= NODES) return;
    int b = node / Ndim, i = node % Ndim;
    float o[9] = {0,0,0, 0,0,0, 0,0,0};
    if (i >= 1 && i <= Ndim - 3) {
        float ca[3][3];
        #pragma unroll
        for (int q = 0; q < 3; q++) {
            const float* p = X + (((size_t)b * Ndim + (i - 1 + q)) * 4 + 1) * 3;
            ca[q][0] = p[0]; ca[q][1] = p[1]; ca[q][2] = p[2];
        }
        float u2[3], u1[3];
        #pragma unroll
        for (int d = 0; d < 3; d++) { u2[d] = ca[1][d]-ca[0][d]; u1[d] = ca[2][d]-ca[1][d]; }
        v_unit(u2); v_unit(u1);
        float n2[3], o1[3], o2[3];
        v_cross(u2, u1, n2); v_unit(n2);
        #pragma unroll
        for (int d = 0; d < 3; d++) o1[d] = u2[d] - u1[d];
        v_unit(o1);
        v_cross(o1, n2, o2);
        o[0]=o1[0]; o[1]=o1[1]; o[2]=o1[2];
        o[3]=n2[0]; o[4]=n2[1]; o[5]=n2[2];
        o[6]=o2[0]; o[7]=o2[1]; o[8]=o2[2];
    }
    #pragma unroll
    for (int q = 0; q < 9; q++) O[(size_t)node * 9 + q] = o[q];
}

// ---------------- edge features: pos-enc(16) | rbf(16) | dU(3) quat(4) ----------------
__global__ void k_edge_feat(const float* __restrict__ X, const float* __restrict__ O,
                            const int* __restrict__ eidx, const float* __restrict__ Dnb,
                            float* __restrict__ Ef)
{
    int e = blockIdx.x * blockDim.x + threadIdx.x;
    if (e >= EDGES) return;
    int node = e / KNN;
    int b = node / Ndim, i = node % Ndim;
    int j = eidx[e];
    float f[EIN];
    float offv = (float)(j - i);
    #pragma unroll
    for (int p = 0; p < 8; p++) {
        float freq = __expf((float)(2 * p) * (-9.210340372f / 16.f));
        float ang = offv * freq;
        f[p] = cosf(ang); f[8 + p] = sinf(ang);
    }
    float D = Dnb[e];
    #pragma unroll
    for (int m = 0; m < 16; m++) {
        float mu = 2.f + 20.f * (float)m / 15.f;
        float z = (D - mu) * (1.f / 1.25f);
        f[16 + m] = __expf(-z * z);
    }
    float Oi[9], Oj[9];
    #pragma unroll
    for (int q = 0; q < 9; q++) { Oi[q] = O[(size_t)node * 9 + q]; Oj[q] = O[(((size_t)b * Ndim + j) * 9) + q]; }
    const float* pi = X + (((size_t)b * Ndim + i) * 4 + 1) * 3;
    const float* pj = X + (((size_t)b * Ndim + j) * 4 + 1) * 3;
    float dX[3] = { pj[0]-pi[0], pj[1]-pi[1], pj[2]-pi[2] };
    float dU[3];
    #pragma unroll
    for (int r = 0; r < 3; r++) dU[r] = Oi[r*3+0]*dX[0] + Oi[r*3+1]*dX[1] + Oi[r*3+2]*dX[2];
    v_unit(dU);
    f[32] = dU[0]; f[33] = dU[1]; f[34] = dU[2];
    float R[9];
    #pragma unroll
    for (int r = 0; r < 3; r++)
        #pragma unroll
        for (int c = 0; c < 3; c++)
            R[r*3+c] = Oi[0*3+r]*Oj[0*3+c] + Oi[1*3+r]*Oj[1*3+c] + Oi[2*3+r]*Oj[2*3+c];
    float Rxx = R[0], Ryy = R[4], Rzz = R[8];
    float m0 = 0.5f * sqrtf(fabsf(1.f + Rxx - Ryy - Rzz) + 1e-6f);
    float m1 = 0.5f * sqrtf(fabsf(1.f - Rxx + Ryy - Rzz) + 1e-6f);
    float m2 = 0.5f * sqrtf(fabsf(1.f - Rxx - Ryy + Rzz) + 1e-6f);
    float s0 = R[7] - R[5], s1 = R[2] - R[6], s2 = R[3] - R[1];
    s0 = (s0>0.f)?1.f:((s0<0.f)?-1.f:0.f);
    s1 = (s1>0.f)?1.f:((s1<0.f)?-1.f:0.f);
    s2 = (s2>0.f)?1.f:((s2<0.f)?-1.f:0.f);
    float wq = 0.5f * sqrtf(fmaxf(1.f + Rxx + Ryy + Rzz, 0.f) + 1e-6f);
    float q[4] = { s0*m0, s1*m1, s2*m2, wq };
    float nq = sqrtf(q[0]*q[0] + q[1]*q[1] + q[2]*q[2] + q[3]*q[3]) + 1e-8f;
    f[35] = q[0]/nq; f[36] = q[1]/nq; f[37] = q[2]/nq; f[38] = q[3]/nq;
    #pragma unroll
    for (int c = 0; c < EIN; c++) Ef[(size_t)e * EIN + c] = f[c];
}

// ---------------- small-K linear + layernorm -> f16 ----------------
__global__ __launch_bounds__(128) void k_embed(const float* __restrict__ F, int fin,
                                               const float* __restrict__ W,
                                               const float* __restrict__ bias,
                                               const float* __restrict__ g,
                                               const float* __restrict__ be,
                                               h16* __restrict__ out16)
{
    __shared__ float red[128];
    __shared__ float fsh[64];
    int row = blockIdx.x, tid = threadIdx.x;
    if (tid < fin) fsh[tid] = F[(size_t)row * fin + tid];
    __syncthreads();
    float y = bias[tid];
    for (int c = 0; c < fin; c++) y += fsh[c] * W[(size_t)c * Hdim + tid];
    red[tid] = y; __syncthreads();
    for (int s = 64; s > 0; s >>= 1) { if (tid < s) red[tid] += red[tid + s]; __syncthreads(); }
    float mu = red[0] * (1.f / 128.f); __syncthreads();
    float d = y - mu;
    red[tid] = d * d; __syncthreads();
    for (int s = 64; s > 0; s >>= 1) { if (tid < s) red[tid] += red[tid + s]; __syncthreads(); }
    float sig = sqrtf(red[0] * (1.f / 128.f) + 1e-6f);
    out16[(size_t)row * Hdim + tid] = (h16)(g[tid] * d / sig + be[tid]);
}

// ---------------- node update: residual (+msg sum) + LN, keep f32 + f16 ----------------
__global__ __launch_bounds__(128) void k_node_update(float* __restrict__ hV32,
                                                     h16* __restrict__ hV16,
                                                     const h16* __restrict__ msg,
                                                     const float* __restrict__ g,
                                                     const float* __restrict__ be,
                                                     int mode)
{
    __shared__ float red[128];
    int node = blockIdx.x, tid = threadIdx.x;
    float acc = 0.f;
    if (mode == 0) {
        const h16* mp = msg + (size_t)node * KNN * Hdim + tid;
        #pragma unroll 5
        for (int k = 0; k < KNN; k++) acc += (float)mp[(size_t)k * Hdim];
        acc *= (1.f / 30.f);
    } else {
        acc = (float)msg[(size_t)node * Hdim + tid];
    }
    float y = hV32[(size_t)node * Hdim + tid] + acc;
    red[tid] = y; __syncthreads();
    for (int s = 64; s > 0; s >>= 1) { if (tid < s) red[tid] += red[tid + s]; __syncthreads(); }
    float mu = red[0] * (1.f / 128.f); __syncthreads();
    float d = y - mu;
    red[tid] = d * d; __syncthreads();
    for (int s = 64; s > 0; s >>= 1) { if (tid < s) red[tid] += red[tid + s]; __syncthreads(); }
    float sig = sqrtf(red[0] * (1.f / 128.f) + 1e-6f);
    float o = g[tid] * d / sig + be[tid];
    hV32[(size_t)node * Hdim + tid] = o;
    hV16[(size_t)node * Hdim + tid] = (h16)o;
}

// ---------------- sequence embedding lookup ----------------
__global__ void k_embed_S(const int* __restrict__ S, const float* __restrict__ Ws,
                          h16* __restrict__ hS)
{
    int t = blockIdx.x * blockDim.x + threadIdx.x;
    if (t >= NODES * Hdim) return;
    int node = t >> 7, h = t & 127;
    hS[t] = (h16)Ws[(size_t)S[node] * Hdim + h];
}

// ---------------- weight convert f32 [K,N] -> f16 transposed [N,K] ----------------
__global__ void k_cvt_wT(const float* __restrict__ W, h16* __restrict__ Wt, int K, int N)
{
    int t = blockIdx.x * blockDim.x + threadIdx.x;
    if (t >= K * N) return;
    int k = t / N, n = t % N;
    Wt[(size_t)n * K + k] = (h16)W[t];
}

// ---------------- output projection + log_softmax ----------------
__global__ __launch_bounds__(32) void k_logits(const float* __restrict__ hV32,
                                               const float* __restrict__ W,
                                               const float* __restrict__ bias,
                                               float* __restrict__ out)
{
    __shared__ float sh[32];
    int node = blockIdx.x, lane = threadIdx.x;
    float l = -3.4e38f;
    if (lane < 21) {
        float acc = bias[lane];
        for (int h = 0; h < Hdim; h++) acc += hV32[(size_t)node * Hdim + h] * W[(size_t)h * 21 + lane];
        l = acc;
    }
    sh[lane] = l; __syncthreads();
    float mx = -3.4e38f;
    for (int q = 0; q < 21; q++) mx = fmaxf(mx, sh[q]);
    __syncthreads();
    sh[lane] = (lane < 21) ? __expf(l - mx) : 0.f;
    __syncthreads();
    float ssum = 0.f;
    for (int q = 0; q < 21; q++) ssum += sh[q];
    if (lane < 21) out[(size_t)node * 21 + lane] = l - mx - logf(ssum);
}

// ---------------- WMMA GEMM with fused h_EV gather + LDS-staged weights ----------------
// A fragment loader. k is 8-half-aligned, chunks never straddle 128-wide segments.
// AKIND 0: plain f16 row-major A[lda]
// AKIND 1: enc h_EV row = [ hV[i] | hE[e] | hV[j] ]                     (Kd=384)
// AKIND 2: dec h_EV row = [ hV[i] | hE[e] | ar?hS[j]:0 | ar?hV[j]:hVenc[j] ] (Kd=512)
template<int AKIND>
__device__ __forceinline__ v8h loadA8(const h16* __restrict__ A, int lda,
                                      const h16* __restrict__ hV, const h16* __restrict__ hE,
                                      const h16* __restrict__ hS, const h16* __restrict__ hVe,
                                      int r, int nodeIdx, int jg, int ar, int k)
{
    if (AKIND == 0) {
        return *(const v8h*)(A + (size_t)r * lda + k);
    } else {
        int seg = k >> 7, off = k & 127;
        if (seg == 0) return *(const v8h*)(hV + (size_t)nodeIdx * Hdim + off);
        if (seg == 1) return *(const v8h*)(hE + (size_t)r * Hdim + off);
        if (AKIND == 1) return *(const v8h*)(hV + (size_t)jg * Hdim + off);
        if (seg == 2) {
            if (ar) return *(const v8h*)(hS + (size_t)jg * Hdim + off);
            v8h z = {}; return z;
        }
        const h16* src = ar ? hV : hVe;
        return *(const v8h*)(src + (size_t)jg * Hdim + off);
    }
}

// 8 waves per block; each wave: 16 rows x 64 cols (4 acc tiles).
// B (weights) slice staged in LDS with 16-byte-granule XOR swizzle (chunk ^ (col&15))
// -> 16 lanes of a ds_load_b128 fragment read hit all 64 banks, no padding, LDS = 64*Kd*2 <= 64KB.
// Steady-state k-loop: unconditional A prefetch (global) + 8 batched ds_loads, then 4 WMMAs
// with staggered dscnt waits; last k-step peeled so the loop body is branch-free.
// M, Nd multiples of 128/64 at every call site -> EXEC all-ones for WMMA.
template<int AKIND>
__global__ __launch_bounds__(256) void k_gemm(const h16* __restrict__ A, int lda,
                                              const h16* __restrict__ hV, const h16* __restrict__ hE,
                                              const h16* __restrict__ hS, const h16* __restrict__ hVe,
                                              const int* __restrict__ eidx,
                                              const h16* __restrict__ Wt,      // [Nd][Kd] f16
                                              const float* __restrict__ bias,
                                              h16* __restrict__ C16, float* __restrict__ C32,
                                              int M, int Nd, int Kd, int relu)
{
    extern __shared__ h16 sB[];                 // 64 cols x Kd halves, swizzled
    int tid  = threadIdx.x;
    int wave = tid >> 5, lane = tid & 31;
    int rowBase = (blockIdx.x * 8 + wave) * 16;
    int colBase = blockIdx.y * 64;
    int lr = lane & 15, hi = lane >> 4;

    // ---- stage weight slice into LDS (cooperative, b128 granules) ----
    int nchunk = Kd >> 3;                       // 16-byte chunks per column
    for (int idx = tid; idx < 64 * nchunk; idx += 256) {
        int col = idx / nchunk, ch = idx % nchunk;
        v8h v = *(const v8h*)(Wt + (size_t)(colBase + col) * Kd + (ch << 3));
        *(v8h*)(sB + (size_t)col * Kd + (size_t)((ch ^ (col & 15)) << 3)) = v;
    }
    __syncthreads();

    int r = rowBase + lr;                       // A row this lane feeds
    int nodeIdx = 0, jg = 0, ar = 0;
    if (AKIND != 0) {
        nodeIdx = r / KNN;                      // global node b*N+i
        int i  = nodeIdx & (Ndim - 1);
        int bb = nodeIdx >> 10;
        int j  = eidx[r];
        jg = (bb << 10) + j;
        ar = (j < i) ? 1 : 0;
    }

    v8f acc[4] = {};

    // one k-step: batch all 8 B ds_loads, then the 4 WMMAs (staggered waits)
    auto kstep = [&](const v8h& alo_, const v8h& ahi_, int kk_) {
        v16h af;
        #pragma unroll
        for (int q = 0; q < 8; q++) { af[q] = alo_[q]; af[q + 8] = ahi_[q]; }
        int ch0 = (kk_ >> 3) + hi;              // chunk index of k0; +2 chunks = +16 halves
        v8h blo[4], bhi[4];
        #pragma unroll
        for (int t = 0; t < 4; t++) {
            int c = t * 16 + lr;                // local column
            const h16* bp = sB + (size_t)c * Kd;
            blo[t] = *(const v8h*)(bp + (size_t)((ch0       ^ (c & 15)) << 3));
            bhi[t] = *(const v8h*)(bp + (size_t)(((ch0 + 2) ^ (c & 15)) << 3));
        }
        #pragma unroll
        for (int t = 0; t < 4; t++) {
            v16h bf;
            #pragma unroll
            for (int q = 0; q < 8; q++) { bf[q] = blo[t][q]; bf[q + 8] = bhi[t][q]; }
            acc[t] = __builtin_amdgcn_wmma_f32_16x16x32_f16(
                false, af, false, bf, (short)0, acc[t], false, false);
        }
    };

    int k0 = hi << 3;                           // lanes 16-31 shifted by 8 per A-layout
    v8h alo = loadA8<AKIND>(A, lda, hV, hE, hS, hVe, r, nodeIdx, jg, ar, k0);
    v8h ahi = loadA8<AKIND>(A, lda, hV, hE, hS, hVe, r, nodeIdx, jg, ar, k0 + 16);

    int kk = 0;
    for (; kk + 32 < Kd; kk += 32) {
        // unconditional prefetch of next A fragment (hides global latency behind WMMA)
        int kn = kk + 32 + (hi << 3);
        v8h nlo = loadA8<AKIND>(A, lda, hV, hE, hS, hVe, r, nodeIdx, jg, ar, kn);
        v8h nhi = loadA8<AKIND>(A, lda, hV, hE, hS, hVe, r, nodeIdx, jg, ar, kn + 16);
        kstep(alo, ahi, kk);
        alo = nlo; ahi = nhi;
    }
    kstep(alo, ahi, kk);                        // peeled last k-step

    int mr0 = rowBase + hi * 8;                 // C layout: VGPR q -> M = q (+8 for lanes 16-31)
    #pragma unroll
    for (int t = 0; t < 4; t++) {
        int c = colBase + t * 16 + lr;
        float bv = bias[c];
        #pragma unroll
        for (int q = 0; q < 8; q++) {
            float v = acc[t][q] + bv;
            if (relu) v = fmaxf(v, 0.f);
            size_t o = (size_t)(mr0 + q) * Nd + c;
            if (C16) C16[o] = (h16)v;
            if (C32) C32[o] = v;
        }
    }
}

// ---------------- host driver ----------------
extern "C" void kernel_launch(void* const* d_in, const int* in_sizes, int n_in,
                              void* d_out, int out_size, void* d_ws, size_t ws_size,
                              hipStream_t stream)
{
    (void)in_sizes; (void)n_in; (void)out_size; (void)ws_size;
    const float* X    = (const float*)d_in[0];
    const int*   S    = (const int*)d_in[1];
    const float* mask = (const float*)d_in[3];
    auto f32 = [&](int i) { return (const float*)d_in[i]; };
    // params flattened DFS in setup_inputs() insertion order, starting at index 8:
    // 8..15 feat, 16/17 W_v, 18/19 W_e, 20 W_s, 21+14*l per layer, 105/106 W_out

    char* wsp = (char*)d_ws; size_t off = 0;
    auto alloc = [&](size_t bytes) -> void* {
        void* p = wsp + off; off += (bytes + 255) & ~(size_t)255; return p;
    };
    float* Vfeat = (float*)alloc((size_t)NODES * 6 * 4);
    float* O9    = (float*)alloc((size_t)NODES * 9 * 4);
    int*   eidx  = (int*)alloc((size_t)EDGES * 4);
    float* Dnb   = (float*)alloc((size_t)EDGES * 4);
    float* Ef    = (float*)alloc((size_t)EDGES * EIN * 4);
    h16*   hE16  = (h16*)alloc((size_t)EDGES * Hdim * 2);
    h16*   hV16  = (h16*)alloc((size_t)NODES * Hdim * 2);
    float* hV32  = (float*)alloc((size_t)NODES * Hdim * 4);
    h16*   hVe16 = (h16*)alloc((size_t)NODES * Hdim * 2);
    h16*   hS16  = (h16*)alloc((size_t)NODES * Hdim * 2);
    h16*   t1    = (h16*)alloc((size_t)EDGES * Hdim * 2);
    h16*   t2    = (h16*)alloc((size_t)EDGES * Hdim * 2);
    h16*   ffn   = (h16*)alloc((size_t)NODES * 512 * 2);
    h16*   dh    = (h16*)alloc((size_t)NODES * Hdim * 2);
    h16*   tV    = (h16*)alloc((size_t)NODES * Hdim * 2);

    // f16 transposed weights
    auto cvt = [&](int pi, int K, int N) -> h16* {
        h16* p = (h16*)alloc((size_t)K * N * 2);
        int tot = K * N;
        k_cvt_wT<<<(tot + 255) / 256, 256, 0, stream>>>(f32(pi), p, K, N);
        return p;
    };
    h16* Wv = cvt(16, 128, 128);
    h16* We = cvt(18, 128, 128);
    h16 *W1w[6], *W2w[6], *W3w[6], *Fiw[6], *Fow[6];
    for (int l = 0; l < 6; l++) {
        int base = 21 + l * 14;
        int k1 = (l < 3) ? 384 : 512;
        W1w[l] = cvt(base + 0, k1, 128);
        W2w[l] = cvt(base + 2, 128, 128);
        W3w[l] = cvt(base + 4, 128, 128);
        Fiw[l] = cvt(base + 10, 128, 512);
        Fow[l] = cvt(base + 12, 512, 128);
    }

    auto gemm0 = [&](const h16* A, int lda, const h16* Wt, const float* bias,
                     h16* C16, float* C32, int M, int Nd, int Kd, int relu) {
        dim3 g(M / 128, Nd / 64);
        size_t shm = (size_t)64 * Kd * 2;
        k_gemm<0><<<g, 256, shm, stream>>>(A, lda, nullptr, nullptr, nullptr, nullptr,
                                           nullptr, Wt, bias, C16, C32, M, Nd, Kd, relu);
    };

    // ---- geometric features ----
    k_dihedrals<<<(NODES * 3 + 255) / 256, 256, 0, stream>>>(X, Vfeat);
    k_topk<<<dim3(Ndim / 128, Bdim), 128, 0, stream>>>(X, mask, eidx, Dnb);
    k_orient<<<(NODES + 255) / 256, 256, 0, stream>>>(X, O9);
    k_edge_feat<<<(EDGES + 255) / 256, 256, 0, stream>>>(X, O9, eidx, Dnb, Ef);

    // ---- embeddings + LN (small K -> VALU), then WMMA projections ----
    k_embed<<<NODES, 128, 0, stream>>>(Vfeat, 6,   f32(8),  f32(9),  f32(12), f32(13), tV);
    k_embed<<<EDGES, 128, 0, stream>>>(Ef,    EIN, f32(10), f32(11), f32(14), f32(15), t1);
    gemm0(tV, 128, Wv, f32(17), hV16, hV32, NODES, 128, 128, 0);    // h_V
    gemm0(t1, 128, We, f32(19), hE16, nullptr, EDGES, 128, 128, 0); // h_E

    k_embed_S<<<(NODES * Hdim + 255) / 256, 256, 0, stream>>>(S, f32(20), hS16);

    // ---- 3 encoder + 3 decoder MPNN layers (mask all-ones: mask factors folded) ----
    for (int l = 0; l < 6; l++) {
        int base = 21 + l * 14;
        if (l < 3) {
            k_gemm<1><<<dim3(EDGES / 128, 2), 256, (size_t)64 * 384 * 2, stream>>>(
                nullptr, 0, hV16, hE16, nullptr, nullptr, eidx,
                W1w[l], f32(base + 1), t1, nullptr, EDGES, 128, 384, 1);
        } else {
            k_gemm<2><<<dim3(EDGES / 128, 2), 256, (size_t)64 * 512 * 2, stream>>>(
                nullptr, 0, hV16, hE16, hS16, hVe16, eidx,
                W1w[l], f32(base + 1), t1, nullptr, EDGES, 128, 512, 1);
        }
        gemm0(t1, 128, W2w[l], f32(base + 3), t2, nullptr, EDGES, 128, 128, 1);
        gemm0(t2, 128, W3w[l], f32(base + 5), t1, nullptr, EDGES, 128, 128, 0);
        k_node_update<<<NODES, 128, 0, stream>>>(hV32, hV16, t1, f32(base + 6), f32(base + 7), 0);
        gemm0(hV16, 128, Fiw[l], f32(base + 11), ffn, nullptr, NODES, 512, 128, 1);
        gemm0(ffn, 512, Fow[l], f32(base + 13), dh, nullptr, NODES, 128, 512, 0);
        k_node_update<<<NODES, 128, 0, stream>>>(hV32, hV16, dh, f32(base + 8), f32(base + 9), 1);
        if (l == 2) {
            // snapshot encoder output for the decoder forward (non-AR) path
            hipMemcpyAsync(hVe16, hV16, (size_t)NODES * Hdim * 2,
                           hipMemcpyDeviceToDevice, stream);
        }
    }

    // ---- output head ----
    k_logits<<<NODES, 32, 0, stream>>>(hV32, f32(105), f32(106), (float*)d_out);
}
